// GRU_28321014350137
// MI455X (gfx1250) — matrix operand order repaired and checked by reading, not compiled, not run on hardware
//
#include <hip/hip_runtime.h>

// ---------------- problem constants ----------------
constexpr int kIn    = 512;               // INPUT_SIZE
constexpr int kHid   = 1024;              // HIDDEN_SIZE
constexpr int kBatch = 64;                // BATCH
constexpr int kSeq   = 512;               // SEQ
constexpr int kNG    = 3 * kHid;          // 3072 packed gate columns (r,z,n)
constexpr int kMRows = kBatch * kSeq;     // 32768 rows of the input GEMM

typedef __bf16 bf16;
typedef __attribute__((ext_vector_type(16))) __bf16 v16bf;
typedef __attribute__((ext_vector_type(8)))  float  v8f;

// ---- CDNA5 async memory->LDS path (GLOBAL_LOAD_ASYNC_TO_LDS_B128, ASYNCcnt) ----
#if defined(__has_builtin)
#if __has_builtin(__builtin_amdgcn_global_load_async_to_lds_b128) && \
    __has_builtin(__builtin_amdgcn_s_wait_asynccnt)
#define USE_ASYNC_LDS 1
#endif
#endif
#ifndef USE_ASYNC_LDS
#define USE_ASYNC_LDS 0
#endif

#if USE_ASYNC_LDS
typedef int vint4 __attribute__((vector_size(4 * sizeof(int))));
__device__ __forceinline__ void async_stage16B(const void* g, void* l) {
    __builtin_amdgcn_global_load_async_to_lds_b128(
        (__attribute__((address_space(1))) vint4*)g,
        (__attribute__((address_space(3))) vint4*)l,
        /*offset=*/0, /*cpol=*/0);
}
#endif

union FragBF {
    v16bf v;
    uint4 q[2];
};

__device__ __forceinline__ v8f wmma_bf16_acc(v16bf a, v16bf b, v8f c) {
    // D = A(16x32 bf16) * B(32x16 bf16) + C(16x16 f32)
    return __builtin_amdgcn_wmma_f32_16x16x32_bf16(
        /*neg_a=*/false, a, /*neg_b=*/false, b,
        /*c_mod=*/(short)0, c, /*reuse_a=*/false, /*reuse_b=*/false);
}

// ---------------- conversion / packing kernels ----------------
__global__ void k_cvt_bf16(const float* __restrict__ src, bf16* __restrict__ dst, int n) {
    int i = blockIdx.x * 256 + threadIdx.x;
    if (i < n) dst[i] = (bf16)src[i];
}

__global__ void k_copy_f32(const float* __restrict__ src, float* __restrict__ dst, int n) {
    int i = blockIdx.x * 256 + threadIdx.x;
    if (i < n) dst[i] = src[i];
}

// ---------------- input projection GEMM ----------------
// Y[row, n] = sum_k X[row,k] * Wp[n,k] + bi[n]
// block tile: 64 rows x 128 cols, 8 waves; wave owns one 16-col strip, 4 M-tiles.
__global__ __launch_bounds__(256) void k_xproj(
    const bf16*  __restrict__ X,    // [kMRows, kIn] bf16 row-major
    const bf16*  __restrict__ Wp,   // [kNG,   kIn] bf16 row-major (gates packed)
    const float* __restrict__ bi,   // [kNG]
    float*       __restrict__ Y)    // [kMRows, kNG]
{
#if USE_ASYNC_LDS
    __shared__ bf16 As[2][64 * 32];                   // double-buffered A tile
#else
    __shared__ bf16 As[1][64 * 32];
#endif

    const int tid  = threadIdx.x;
    const int wave = tid >> 5;
    const int lane = tid & 31;
    const int bn   = blockIdx.x;                      // 24 col blocks of 128
    const int bm   = blockIdx.y;                      // 512 row blocks of 64

    const int ncol = bn * 128 + wave * 16 + (lane & 15);
    const int hiA  = (lane & 16) ? 8  : 0;            // A: upper lane-half -> +8 elems
    const int hiB  = (lane & 16) ? 8  : 0;            // B: upper lane-half -> +8 elems

    const int r = tid >> 2, c = (tid & 3) << 3;       // this thread's 16B staging slot
    const bf16* xsrc   = X + (size_t)(bm * 64 + r) * kIn + c;
    const bf16* wbase  = Wp + (size_t)ncol * kIn + hiB;
    constexpr int NK = kIn / 32;

    v8f acc[4] = {};

#if USE_ASYNC_LDS
    async_stage16B(xsrc, &As[0][r * 32 + c]);
    for (int kt = 0; kt < NK; ++kt) {
        __builtin_amdgcn_s_wait_asynccnt(0);          // my slice of tile kt is in LDS
        __syncthreads();                              // everyone's slice is in LDS
        if (kt + 1 < NK)                              // prefetch tile kt+1 while computing
            async_stage16B(xsrc + (kt + 1) * 32, &As[(kt + 1) & 1][r * 32 + c]);

        FragBF bf;
        const bf16* wb = wbase + kt * 32;
        bf.q[0] = *(const uint4*)(wb);
        bf.q[1] = *(const uint4*)(wb + 16);
        const bf16* curA = As[kt & 1];
#pragma unroll
        for (int mt = 0; mt < 4; ++mt) {
            FragBF af;
            const bf16* ab = &curA[(mt * 16 + (lane & 15)) * 32 + hiA];
            af.q[0] = *(const uint4*)(ab);
            af.q[1] = *(const uint4*)(ab + 16);
            acc[mt] = wmma_bf16_acc(af.v, bf.v, acc[mt]);
        }
    }
#else
    for (int kt = 0; kt < NK; ++kt) {
        *(uint4*)&As[0][r * 32 + c] = *(const uint4*)(xsrc + kt * 32);
        __syncthreads();

        FragBF bf;
        const bf16* wb = wbase + kt * 32;
        bf.q[0] = *(const uint4*)(wb);
        bf.q[1] = *(const uint4*)(wb + 16);
#pragma unroll
        for (int mt = 0; mt < 4; ++mt) {
            FragBF af;
            const bf16* ab = &As[0][(mt * 16 + (lane & 15)) * 32 + hiA];
            af.q[0] = *(const uint4*)(ab);
            af.q[1] = *(const uint4*)(ab + 16);
            acc[mt] = wmma_bf16_acc(af.v, bf.v, acc[mt]);
        }
        __syncthreads();
    }
#endif

    const float bv = bi[ncol];
#pragma unroll
    for (int mt = 0; mt < 4; ++mt) {
        const int m0 = bm * 64 + mt * 16 + ((lane >> 4) << 3);
#pragma unroll
        for (int i = 0; i < 8; ++i)
            Y[(size_t)(m0 + i) * kNG + ncol] = acc[mt][i] + bv;
    }
}

// ---------------- one recurrent GRU timestep ----------------
// 16 blocks, each owns 64 hidden columns; computes r,z,n projections for its
// columns (3 gates x 64x64, K=1024), then the fused elementwise update.
__global__ __launch_bounds__(256) void k_gru_step(
    const bf16*  __restrict__ hbf_in,   // [kBatch, kHid] h_{t-1} in bf16 (A matrix)
    const float* __restrict__ hf_in,    // [kBatch, kHid] h_{t-1} in f32
    bf16*        __restrict__ hbf_out,
    float*       __restrict__ hf_out,
    const bf16*  __restrict__ Wh,       // [kNG, kHid] bf16, gates packed r,z,n
    const float* __restrict__ bh,       // [kNG]
    const float* __restrict__ xacc,     // [kMRows, kNG] precomputed x-projections
    float*       __restrict__ out,      // [kBatch, kSeq, kHid]
    float*       __restrict__ hlast,    // [kBatch, kHid]
    int t)
{
#if USE_ASYNC_LDS
    __shared__ bf16  Hs[2][64 * 32];       // double-buffered h staging tile
#else
    __shared__ bf16  Hs[1][64 * 32];
#endif
    __shared__ float Gs[3 * 64 * 64];      // 48 KB gate accumulators

    const int tid  = threadIdx.x;
    const int wave = tid >> 5;
    const int lane = tid & 31;
    const int nblk = blockIdx.x;                 // 16 column blocks of 64

    const int mt    = wave >> 1;                 // wave's M-tile (0..3)
    const int jbase = (wave & 1) * 6;            // 12 (gate,ntile) jobs over wave pairs
    const int hiA   = (lane & 16) ? 8  : 0;
    const int hiB   = (lane & 16) ? 8  : 0;

    const int r = tid >> 2, c = (tid & 3) << 3;
    const bf16* hsrc = hbf_in + (size_t)r * kHid + c;
    constexpr int NK = kHid / 32;

    v8f acc[6] = {};

#if USE_ASYNC_LDS
    async_stage16B(hsrc, &Hs[0][r * 32 + c]);
    for (int kt = 0; kt < NK; ++kt) {
        __builtin_amdgcn_s_wait_asynccnt(0);
        __syncthreads();
        if (kt + 1 < NK)
            async_stage16B(hsrc + (kt + 1) * 32, &Hs[(kt + 1) & 1][r * 32 + c]);

        const int k0 = kt * 32;
        FragBF af;
        const bf16* ab = &Hs[kt & 1][(mt * 16 + (lane & 15)) * 32 + hiA];
        af.q[0] = *(const uint4*)(ab);
        af.q[1] = *(const uint4*)(ab + 16);
#pragma unroll
        for (int jj = 0; jj < 6; ++jj) {
            const int j    = jbase + jj;
            const int ncol = (j >> 2) * kHid + nblk * 64 + (j & 3) * 16 + (lane & 15);
            FragBF bf;
            const bf16* wb = Wh + (size_t)ncol * kHid + k0 + hiB;
            bf.q[0] = *(const uint4*)(wb);
            bf.q[1] = *(const uint4*)(wb + 16);
            acc[jj] = wmma_bf16_acc(af.v, bf.v, acc[jj]);
        }
    }
#else
    for (int kt = 0; kt < NK; ++kt) {
        const int k0 = kt * 32;
        *(uint4*)&Hs[0][r * 32 + c] = *(const uint4*)(hsrc + k0);
        __syncthreads();

        FragBF af;
        const bf16* ab = &Hs[0][(mt * 16 + (lane & 15)) * 32 + hiA];
        af.q[0] = *(const uint4*)(ab);
        af.q[1] = *(const uint4*)(ab + 16);
#pragma unroll
        for (int jj = 0; jj < 6; ++jj) {
            const int j    = jbase + jj;
            const int ncol = (j >> 2) * kHid + nblk * 64 + (j & 3) * 16 + (lane & 15);
            FragBF bf;
            const bf16* wb = Wh + (size_t)ncol * kHid + k0 + hiB;
            bf.q[0] = *(const uint4*)(wb);
            bf.q[1] = *(const uint4*)(wb + 16);
            acc[jj] = wmma_bf16_acc(af.v, bf.v, acc[jj]);
        }
        __syncthreads();
    }
#endif

    // scatter gate accumulators to LDS: Gs[gate][m][n_local]
    __syncthreads();
#pragma unroll
    for (int jj = 0; jj < 6; ++jj) {
        const int j  = jbase + jj;
        const int nl = (j & 3) * 16 + (lane & 15);
        const int m0 = mt * 16 + ((lane >> 4) << 3);
#pragma unroll
        for (int i = 0; i < 8; ++i)
            Gs[(j >> 2) * 4096 + (m0 + i) * 64 + nl] = acc[jj][i];
    }
    __syncthreads();

    // fused GRU elementwise update on the 64x64 block
    for (int e = tid; e < 4096; e += 256) {
        const int m  = e >> 6;
        const int nl = e & 63;
        const int h  = nblk * 64 + nl;
        const size_t xrow = ((size_t)m * kSeq + t) * kNG;

        const float ar = Gs[e]        + bh[h];
        const float az = Gs[4096 + e] + bh[kHid + h];
        const float an = Gs[8192 + e] + bh[2 * kHid + h];

        const float r  = 1.0f / (1.0f + __expf(-(xacc[xrow + h]            + ar)));
        const float z  = 1.0f / (1.0f + __expf(-(xacc[xrow + kHid + h]     + az)));
        const float nn = tanhf(xacc[xrow + 2 * kHid + h] + r * an);

        const float hp = hf_in[m * kHid + h];
        const float hv = (1.0f - z) * nn + z * hp;

        out[((size_t)m * kSeq + t) * kHid + h] = hv;
        hf_out[m * kHid + h]  = hv;
        hbf_out[m * kHid + h] = (bf16)hv;
        if (t == kSeq - 1) hlast[m * kHid + h] = hv;
    }
}

// ---------------- host launch ----------------
extern "C" void kernel_launch(void* const* d_in, const int* in_sizes, int n_in,
                              void* d_out, int out_size, void* d_ws, size_t ws_size,
                              hipStream_t stream) {
    const float* inputs = (const float*)d_in[0];
    const float* h0     = (const float*)d_in[1];
    const float* w_i[3] = {(const float*)d_in[2], (const float*)d_in[3], (const float*)d_in[4]};
    const float* b_i[3] = {(const float*)d_in[5], (const float*)d_in[6], (const float*)d_in[7]};
    const float* w_h[3] = {(const float*)d_in[8], (const float*)d_in[9], (const float*)d_in[10]};
    const float* b_h[3] = {(const float*)d_in[11], (const float*)d_in[12], (const float*)d_in[13]};

    char* ws = (char*)d_ws;
    size_t off = 0;
    auto alloc = [&](size_t bytes) -> char* {
        char* p = ws + off;
        off = (off + bytes + 255) & ~(size_t)255;
        return p;
    };

    bf16*  in_bf = (bf16*) alloc((size_t)kMRows * kIn * sizeof(bf16));
    bf16*  wi_bf = (bf16*) alloc((size_t)kNG * kIn * sizeof(bf16));
    bf16*  wh_bf = (bf16*) alloc((size_t)kNG * kHid * sizeof(bf16));
    float* bi_p  = (float*)alloc((size_t)kNG * sizeof(float));
    float* bh_p  = (float*)alloc((size_t)kNG * sizeof(float));
    bf16*  hbf0  = (bf16*) alloc((size_t)kBatch * kHid * sizeof(bf16));
    bf16*  hbf1  = (bf16*) alloc((size_t)kBatch * kHid * sizeof(bf16));
    float* hf0   = (float*)alloc((size_t)kBatch * kHid * sizeof(float));
    float* hf1   = (float*)alloc((size_t)kBatch * kHid * sizeof(float));
    float* xacc  = (float*)alloc((size_t)kMRows * kNG * sizeof(float));

    // --- convert inputs to bf16, pack gate weights/biases (r,z,n order) ---
    k_cvt_bf16<<<(kMRows * kIn) / 256, 256, 0, stream>>>(inputs, in_bf, kMRows * kIn);
    for (int g = 0; g < 3; ++g) {
        k_cvt_bf16<<<(kHid * kIn) / 256, 256, 0, stream>>>(
            w_i[g], wi_bf + (size_t)g * kHid * kIn, kHid * kIn);
        k_cvt_bf16<<<(kHid * kHid) / 256, 256, 0, stream>>>(
            w_h[g], wh_bf + (size_t)g * kHid * kHid, kHid * kHid);
        k_copy_f32<<<kHid / 256, 256, 0, stream>>>(b_i[g], bi_p + g * kHid, kHid);
        k_copy_f32<<<kHid / 256, 256, 0, stream>>>(b_h[g], bh_p + g * kHid, kHid);
    }
    k_cvt_bf16<<<(kBatch * kHid) / 256, 256, 0, stream>>>(h0, hbf0, kBatch * kHid);
    k_copy_f32<<<(kBatch * kHid) / 256, 256, 0, stream>>>(h0, hf0, kBatch * kHid);

    // --- big input-projection GEMM: xacc = X * Wi^T + bi ---
    dim3 gx(kNG / 128, kMRows / 64);
    k_xproj<<<gx, 256, 0, stream>>>(in_bf, wi_bf, bi_p, xacc);

    // --- sequential recurrence: 512 small WMMA GEMM + fused-update launches ---
    float* out   = (float*)d_out;
    float* hlast = out + (size_t)kBatch * kSeq * kHid;
    bf16*  hbf[2] = {hbf0, hbf1};
    float* hf[2]  = {hf0, hf1};
    for (int t = 0; t < kSeq; ++t) {
        const int cur = t & 1, nxt = cur ^ 1;
        k_gru_step<<<kHid / 64, 256, 0, stream>>>(
            hbf[cur], hf[cur], hbf[nxt], hf[nxt],
            wh_bf, bh_p, xacc, out, hlast, t);
    }
}
